// BiLSTM_W2V_CRF_41223096107433
// MI455X (gfx1250) — compile-verified, matrix operand order checked by reading, not run on hardware
//
#include <hip/hip_runtime.h>
#include <hip/hip_bf16.h>

typedef __attribute__((ext_vector_type(16))) __bf16   v16bf;
typedef __attribute__((ext_vector_type(8)))  float    v8f;
typedef __attribute__((ext_vector_type(4)))  float    v4f;
typedef __attribute__((ext_vector_type(4)))  unsigned v4u;
typedef __attribute__((ext_vector_type(8)))  int      v8i;
typedef __attribute__((ext_vector_type(4)))  int      v4i;

union Frag  { v16bf v; v4u q[2]; };          // 16x32 bf16 A/B fragment
union AccU  { v8f   v; v4f q[2]; };          // 16x16 f32 C/D fragment
union Pack8 { __bf16 h[8]; v4u q; };

#define SEQ   512
#define BATCH 64
#define EMB   300
#define EMBP  320      // padded K
#define HD    128
#define GATES 512      // 4*HD
#define NTAGS 9

__device__ __forceinline__ float sigm(float x) { return 1.0f / (1.0f + __expf(-x)); }

// ---------------------------------------------------------------------------
// Pack fp32 [rows][K] -> bf16 [rows][Kp], zero padded. 8 elements / thread,
// branch-free (clamped index + value select), b128 stores.
// ---------------------------------------------------------------------------
__global__ void pack_bf16_kernel(const float* __restrict__ src,
                                 __bf16* __restrict__ dst,
                                 int rows, int K, int Kp) {
    const int chunks = Kp >> 3;
    const int idx = blockIdx.x * 256 + threadIdx.x;
    if (idx >= rows * chunks) return;
    const int row = idx / chunks;
    const int k0  = (idx - row * chunks) << 3;
    const float* sr = src + (size_t)row * K;
    Pack8 p;
#pragma unroll
    for (int i = 0; i < 8; ++i) {
        const int k = k0 + i;
        const float f = sr[k < K ? k : (K - 1)];   // in-bounds clamp, no branch
        p.h[i] = (__bf16)(k < K ? f : 0.f);
    }
    *(v4u*)(dst + (size_t)row * Kp + k0) = p.q;
}

// ---------------------------------------------------------------------------
// Kernel 1: G = embed-gather @ Wih^T + (bih+bhh), bf16 WMMA, fp32 acc.
// M = S*B = 32768, N = 512, K = 320 (pre-padded).  Output in WMMA-fragment
// layout: G[dir][s][bt(4)][ntile(32)][lane(32)][8].
// ---------------------------------------------------------------------------
__global__ void input_proj_kernel(const int* __restrict__ words,
                                  const __bf16* __restrict__ emb_bf,
                                  const __bf16* __restrict__ wih_bf,
                                  const float* __restrict__ bih_f,
                                  const float* __restrict__ bhh_f,
                                  const float* __restrict__ bih_b,
                                  const float* __restrict__ bhh_b,
                                  float* __restrict__ G) {
    const int dir = blockIdx.y;
    const float* bih = dir ? bih_b : bih_f;
    const float* bhh = dir ? bhh_b : bhh_f;

    const int wave   = threadIdx.x >> 5;
    const int lane   = threadIdx.x & 31;
    const int wg     = blockIdx.x * 8 + wave;
    const int tile_m = wg >> 5;              // 0..2047 -> (s, bt)
    const int tile_n = wg & 31;              // 0..31
    const int half   = lane >> 4;
    const int nl     = lane & 15;
    const int s      = tile_m >> 2;
    const int bt     = tile_m & 3;

    // A-fragment row: M = lane%16 -> batch b = bt*16+nl at time s
    const int word = words[(bt * 16 + nl) * SEQ + s];
    const __bf16* arow = emb_bf + (size_t)word * EMBP;

    // B-fragment column: N = lane%16 ; B(k,n) = Wih[n][k]
    const int n = tile_n * 16 + nl;
    const __bf16* brow = wih_bf + ((size_t)dir * GATES + n) * EMBP;

    AccU acc;
    acc.v = (v8f){0.f, 0.f, 0.f, 0.f, 0.f, 0.f, 0.f, 0.f};

#pragma unroll
    for (int kk = 0; kk < 10; ++kk) {        // 10 * 32 = 320
        Frag A, B;
        const __bf16* ap = arow + kk * 32 + half * 8;
        const __bf16* bp = brow + kk * 32 + half * 8;
        A.q[0] = *(const v4u*)ap;            // K = half*8 + 0..7
        A.q[1] = *(const v4u*)(ap + 16);     // K = 16 + half*8 + 0..7
        B.q[0] = *(const v4u*)bp;
        B.q[1] = *(const v4u*)(bp + 16);
        acc.v = __builtin_amdgcn_wmma_f32_16x16x32_bf16(false, A.v, false, B.v,
                                                        (short)0, acc.v, false, false);
    }

    const float bias = bih[n] + bhh[n];
#pragma unroll
    for (int rr = 0; rr < 8; ++rr) acc.v[rr] += bias;

    float* gp = G + ((((size_t)dir * SEQ + s) * 4 + bt) * 32 + tile_n) * 256 +
                (size_t)lane * 8;
    *(v4f*)gp       = acc.q[0];
    *(v4f*)(gp + 4) = acc.q[1];
}

// ---------------------------------------------------------------------------
// Kernel 2: sequential LSTM scan. 8 blocks = (2 dirs) x (4 batch tiles).
// Whh bf16 (128KB) pulled into LDS by the Tensor Data Mover; h-state in LDS;
// c-state in registers.  Wave w owns N-tiles {w, w+8, w+16, w+24} so the
// i/f/g/o gates of channel w*16+nl land in the same thread.
// ---------------------------------------------------------------------------
__global__ void lstm_scan_kernel(const __bf16* __restrict__ whh_bf,
                                 const float* __restrict__ G,
                                 float* __restrict__ h_out) {
    __shared__ __bf16 whh_s[GATES * HD];     // 128 KB of the 320 KB WGP LDS
    __shared__ __bf16 h_s[16 * HD];          // 4 KB

    const int dir = blockIdx.x >> 2;
    const int bt  = blockIdx.x & 3;
    const int b0  = bt * 16;
    const __bf16* whh_src = whh_bf + (size_t)dir * GATES * HD;
    const float*  Gd = G + (size_t)dir * ((size_t)SEQ * 4 * 32 * 256);

    const int t = threadIdx.x;
    const int wave = t >> 5, lane = t & 31;
    const int half = lane >> 4, nl = lane & 15;
    const int ch = wave * 16 + nl;

    // --- TDM: async tensor load of Whh (2D: 128 x 512 rows of 2B elems) ---
    if (t == 0) {
        const unsigned lds_addr = (unsigned)(size_t)(&whh_s[0]);
        const unsigned long long ga = (unsigned long long)(const void*)whh_src;
        v4u g0; v8i g1; v4i g2, g3; v8i g4;
        g0[0] = 1u;                                   // count=1, no gather
        g0[1] = lds_addr;                             // lds_addr [63:32]
        g0[2] = (unsigned)ga;                         // global_addr [95:64]
        g0[3] = (unsigned)((ga >> 32) & 0x01FFFFFFu) | (2u << 30);  // addr hi | type=2
        g1[0] = (int)(1u << 16);                      // data_size=1 (2 bytes)
        g1[1] = (int)((unsigned)HD << 16);            // tensor_dim0[15:0]
        g1[2] = (int)((unsigned)GATES << 16);         // tensor_dim1[15:0]
        g1[3] = (int)((unsigned)HD << 16);            // tile_dim0
        g1[4] = (int)GATES;                           // tile_dim1
        g1[5] = (int)HD;                              // tensor_dim0_stride lo
        g1[6] = 0;
        g1[7] = 0;
        g2 = (v4i){0, 0, 0, 0};
        g3 = (v4i){0, 0, 0, 0};
        g4 = (v8i){0, 0, 0, 0, 0, 0, 0, 0};
        __builtin_amdgcn_tensor_load_to_lds(g0, g1, g2, g3, g4, 0);
        __builtin_amdgcn_s_wait_tensorcnt((short)0);
    }
    for (int i = t; i < 16 * HD; i += 256) h_s[i] = (__bf16)0.f;

    float c[8];
#pragma unroll
    for (int rr = 0; rr < 8; ++rr) c[rr] = 0.f;

    __syncthreads();   // whh_s + h_s ready

    for (int step = 0; step < SEQ; ++step) {
        const int tt = dir ? (SEQ - 1 - step) : step;

        // C init from fragment-layout G: 2 x global_load_b128 per gate
        AccU acc[4];
        const float* gbase = Gd + (((size_t)tt * 4 + bt) * 32) * 256 + (size_t)lane * 8;
#pragma unroll
        for (int gI = 0; gI < 4; ++gI) {
            const float* gp = gbase + (size_t)(gI * 8 + wave) * 256;
            acc[gI].q[0] = *(const v4f*)gp;
            acc[gI].q[1] = *(const v4f*)(gp + 4);
        }

        if (step + 1 < SEQ) {  // global_prefetch for next step's G tiles
            const int tn = dir ? (SEQ - 2 - step) : (step + 1);
            const float* gnext = Gd + (((size_t)tn * 4 + bt) * 32) * 256 + (size_t)lane * 8;
#pragma unroll
            for (int gI = 0; gI < 4; ++gI)
                __builtin_prefetch(gnext + (size_t)(gI * 8 + wave) * 256, 0, 1);
        }

        // gates += h_prev @ Whh^T : K=128 -> 4 bf16 k-steps, ds_load_b128 frags
#pragma unroll
        for (int kk = 0; kk < 4; ++kk) {
            Frag A;
            const __bf16* ap = h_s + nl * HD + kk * 32 + half * 8;
            A.q[0] = *(const v4u*)ap;
            A.q[1] = *(const v4u*)(ap + 16);
#pragma unroll
            for (int gI = 0; gI < 4; ++gI) {
                Frag B;
                const __bf16* bp = whh_s + (size_t)(gI * HD + ch) * HD + kk * 32 + half * 8;
                B.q[0] = *(const v4u*)bp;
                B.q[1] = *(const v4u*)(bp + 16);
                acc[gI].v = __builtin_amdgcn_wmma_f32_16x16x32_bf16(
                    false, A.v, false, B.v, (short)0, acc[gI].v, false, false);
            }
        }

        __syncthreads();   // all waves done reading h_s

        // LSTM cell update: all four gates of (row, ch) are local to this thread
#pragma unroll
        for (int rr = 0; rr < 8; ++rr) {
            const int row = rr + half * 8;
            const float gi = acc[0].v[rr], gf = acc[1].v[rr];
            const float gg = acc[2].v[rr], go = acc[3].v[rr];
            const float cn = sigm(gf) * c[rr] + sigm(gi) * tanhf(gg);
            c[rr] = cn;
            const float h = sigm(go) * tanhf(cn);
            h_s[row * HD + ch] = (__bf16)h;
            h_out[((size_t)tt * BATCH + b0 + row) * (2 * HD) + dir * HD + ch] = h;
        }

        __syncthreads();   // h_s visible for next step
    }
}

// ---------------------------------------------------------------------------
// Kernel 3: emissions em[s][b][t] = h[s][b][:256] . W_out[t] + b_out[t]
// ---------------------------------------------------------------------------
__global__ void emission_kernel(const float* __restrict__ h,
                                const float* __restrict__ W_out,
                                const float* __restrict__ b_out,
                                float* __restrict__ em) {
    const int idx = blockIdx.x * 256 + threadIdx.x;
    if (idx >= SEQ * BATCH * NTAGS) return;
    const int tag = idx % NTAGS;
    const int sb  = idx / NTAGS;
    const v4f* hr = (const v4f*)(h + (size_t)sb * (2 * HD));
    const v4f* wr = (const v4f*)(W_out + (size_t)tag * (2 * HD));
    float sum = b_out[tag];
#pragma unroll 8
    for (int k = 0; k < (2 * HD) / 4; ++k) {
        const v4f a = hr[k], b = wr[k];
        sum += a[0] * b[0] + a[1] * b[1] + a[2] * b[2] + a[3] * b[3];
    }
    em[idx] = sum;
}

// ---------------------------------------------------------------------------
// Kernel 4: CRF NLL. One thread per batch element.
// ---------------------------------------------------------------------------
__global__ void crf_kernel(const float* __restrict__ em,
                           const int* __restrict__ tags,
                           const unsigned char* __restrict__ mask,
                           const float* __restrict__ start,
                           const float* __restrict__ endt,
                           const float* __restrict__ trans,
                           float* __restrict__ out) {
    __shared__ float tr[NTAGS * NTAGS];
    __shared__ float red[BATCH];
    const int b = threadIdx.x;
    for (int i = b; i < NTAGS * NTAGS; i += BATCH) tr[i] = trans[i];
    __syncthreads();

    const int* tb = tags + b * SEQ;
    const unsigned char* mb = mask + b * SEQ;

    const int t0 = tb[0];
    float score = start[t0] + em[(size_t)b * NTAGS + t0];
    int prev = t0, last = t0;
    for (int s = 1; s < SEQ; ++s) {
        const int tg = tb[s];
        const float m = mb[s] ? 1.f : 0.f;
        score += (tr[prev * NTAGS + tg] + em[((size_t)s * BATCH + b) * NTAGS + tg]) * m;
        if (mb[s]) last = tg;
        prev = tg;
    }
    score += endt[last];

    float alpha[NTAGS];
#pragma unroll
    for (int j = 0; j < NTAGS; ++j) alpha[j] = start[j] + em[(size_t)b * NTAGS + j];
    for (int s = 1; s < SEQ; ++s) {
        const float* ems = em + ((size_t)s * BATCH + b) * NTAGS;
        float na[NTAGS];
#pragma unroll
        for (int j = 0; j < NTAGS; ++j) {
            float mx = -1e30f;
#pragma unroll
            for (int i = 0; i < NTAGS; ++i) mx = fmaxf(mx, alpha[i] + tr[i * NTAGS + j]);
            float ss = 0.f;
#pragma unroll
            for (int i = 0; i < NTAGS; ++i) ss += __expf(alpha[i] + tr[i * NTAGS + j] - mx);
            na[j] = mx + __logf(ss) + ems[j];
        }
        const bool m = mb[s] != 0;
#pragma unroll
        for (int j = 0; j < NTAGS; ++j) alpha[j] = m ? na[j] : alpha[j];
    }
    float mx = -1e30f;
#pragma unroll
    for (int j = 0; j < NTAGS; ++j) mx = fmaxf(mx, alpha[j] + endt[j]);
    float ss = 0.f;
#pragma unroll
    for (int j = 0; j < NTAGS; ++j) ss += __expf(alpha[j] + endt[j] - mx);
    const float denom = mx + __logf(ss);

    red[b] = score - denom;
    __syncthreads();
    if (b == 0) {
        float a = 0.f;
        for (int i = 0; i < BATCH; ++i) a += red[i];
        out[0] = -a / (float)BATCH;
    }
}

// ---------------------------------------------------------------------------
extern "C" void kernel_launch(void* const* d_in, const int* in_sizes, int n_in,
                              void* d_out, int out_size, void* d_ws, size_t ws_size,
                              hipStream_t stream) {
    const int*   words = (const int*)d_in[0];
    const int*   tags  = (const int*)d_in[1];
    const unsigned char* mask = (const unsigned char*)d_in[2];
    const float* emb   = (const float*)d_in[3];
    const float* Wih_f = (const float*)d_in[4];
    const float* Whh_f = (const float*)d_in[5];
    const float* bih_f = (const float*)d_in[6];
    const float* bhh_f = (const float*)d_in[7];
    const float* Wih_b = (const float*)d_in[8];
    const float* Whh_b = (const float*)d_in[9];
    const float* bih_b = (const float*)d_in[10];
    const float* bhh_b = (const float*)d_in[11];
    const float* W_out = (const float*)d_in[12];
    const float* b_out = (const float*)d_in[13];
    const float* start_trans = (const float*)d_in[14];
    const float* end_trans   = (const float*)d_in[15];
    const float* trans       = (const float*)d_in[16];

    // ---- workspace layout (all 16B aligned) ----
    char* W = (char*)d_ws;
    float* G  = (float*)W;                                   // 2*512*4*32*256 f32 = 128 MB
    const size_t G_bytes = 2ull * SEQ * 4 * 32 * 256 * 4;
    float* h  = (float*)(W + G_bytes);                       // 512*64*256 f32 = 32 MB
    const size_t h_bytes = (size_t)SEQ * BATCH * 2 * HD * 4;
    float* em = (float*)(W + G_bytes + h_bytes);             // 512*64*9 f32
    const size_t em_bytes = (size_t)SEQ * BATCH * NTAGS * 4;
    __bf16* emb_bf = (__bf16*)(W + G_bytes + h_bytes + em_bytes);      // 100000*320 bf16
    const size_t emb_bytes = 100000ull * EMBP * 2;
    __bf16* wih_bf = (__bf16*)((char*)emb_bf + emb_bytes);             // 2*512*320 bf16
    const size_t wih_bytes = 2ull * GATES * EMBP * 2;
    __bf16* whh_bf = (__bf16*)((char*)wih_bf + wih_bytes);             // 2*512*128 bf16

    // 0) one-shot bf16 packing (zero-padded K)
    pack_bf16_kernel<<<(100000 * (EMBP / 8) + 255) / 256, 256, 0, stream>>>(
        emb, emb_bf, 100000, EMB, EMBP);
    pack_bf16_kernel<<<(GATES * (EMBP / 8) + 255) / 256, 256, 0, stream>>>(
        Wih_f, wih_bf, GATES, EMB, EMBP);
    pack_bf16_kernel<<<(GATES * (EMBP / 8) + 255) / 256, 256, 0, stream>>>(
        Wih_b, wih_bf + (size_t)GATES * EMBP, GATES, EMB, EMBP);
    pack_bf16_kernel<<<(GATES * (HD / 8) + 255) / 256, 256, 0, stream>>>(
        Whh_f, whh_bf, GATES, HD, HD);
    pack_bf16_kernel<<<(GATES * (HD / 8) + 255) / 256, 256, 0, stream>>>(
        Whh_b, whh_bf + (size_t)GATES * HD, GATES, HD, HD);

    // 1) embed-gather + input projection GEMM (both directions)
    input_proj_kernel<<<dim3(8192, 2), 256, 0, stream>>>(
        words, emb_bf, wih_bf, bih_f, bhh_f, bih_b, bhh_b, G);

    // 2) sequential BiLSTM scan: 2 dirs x 4 batch tiles
    lstm_scan_kernel<<<8, 256, 0, stream>>>(whh_bf, G, h);

    // 3) emissions
    emission_kernel<<<(SEQ * BATCH * NTAGS + 255) / 256, 256, 0, stream>>>(
        h, W_out, b_out, em);

    // 4) CRF NLL -> scalar
    crf_kernel<<<1, BATCH, 0, stream>>>(em, tags, mask, start_trans, end_trans,
                                        trans, (float*)d_out);
}
